// HiddenBlock_90529320665929
// MI455X (gfx1250) — matrix operand order repaired and checked by reading, not compiled, not run on hardware
//
#include <hip/hip_runtime.h>
#include <hip/hip_bf16.h>
#include <math.h>

// ---------------------------------------------------------------------------
// Transformer block (pre-LN attn + MLP, causal) for gfx1250 (MI455X, wave32).
// GEMMs: v_wmma_f32_16x16x32_f16, tiles staged via global_load_async_to_lds
// (double-buffered, ASYNCcnt-synchronized). Residual stream + LN stats fp32.
// ---------------------------------------------------------------------------

typedef __attribute__((ext_vector_type(16))) _Float16 v16h;
typedef __attribute__((ext_vector_type(8)))  float    v8f;

union Frag16 { v16h v; unsigned u[8]; };
union HPack  { _Float16 h[2]; unsigned u; };

__device__ __forceinline__ v8f wmma_f16(v16h a, v16h b, v8f c) {
  // (neg_a, A, neg_b, B, c_mod, C, reuse_a, reuse_b)
  return __builtin_amdgcn_wmma_f32_16x16x32_f16(false, a, false, b, (short)0, c,
                                                false, false);
}

// K-pair base for the 16-bit A-matrix 16x32 fragment (contiguous k order):
// lanes 0-15 : VGPR0-7 -> K=0..15
// lanes 16-31: VGPR0-7 -> K=16..31
__device__ __forceinline__ int aklo(int i, int lane) {
  return 2 * i + ((lane < 16) ? 0 : 16);
}

// Per-lane async DMA of 16 bytes global -> LDS (tracked by ASYNCcnt).
__device__ __forceinline__ void async_b128(unsigned lds_off, const void* gaddr) {
  asm volatile("global_load_async_to_lds_b128 %0, %1, off"
               :: "v"(lds_off), "v"(gaddr) : "memory");
}

// ---------------------------------------------------------------------------
// Weight fp32 -> f16, re-laid out for WMMA B-fragments:
// out dword index ((k/32)*N + n)*16 + pk  holds halves (W[k/32*32+2pk][n],
// W[..+2pk+1][n]).  A 32xK-step, 256-col tile is then 16 KB *contiguous*.
// ---------------------------------------------------------------------------
__global__ void pack_weight_kernel(const float* __restrict__ in,
                                   unsigned* __restrict__ out, int K, int N) {
  const int total = (K >> 1) * N;  // output dwords
  int i = blockIdx.x * blockDim.x + threadIdx.x;
  const int stride = gridDim.x * blockDim.x;
  for (; i < total; i += stride) {
    const int c = i / (N * 16);
    const int rem = i - c * (N * 16);
    const int n = rem >> 4;
    const int pk = rem & 15;
    const int k = c * 32 + 2 * pk;
    HPack p;
    p.h[0] = (_Float16)in[(size_t)k * N + n];
    p.h[1] = (_Float16)in[(size_t)(k + 1) * N + n];
    out[i] = p.u;
  }
}

// ---------------------------------------------------------------------------
// LayerNorm over C=1024, fp32 in, f16 out. One 256-thread block per row.
// ---------------------------------------------------------------------------
__global__ __launch_bounds__(256)
void layernorm_f16_kernel(const float* __restrict__ x,
                          const float* __restrict__ w,
                          const float* __restrict__ b,
                          _Float16* __restrict__ out) {
  __shared__ float red[256];
  const int t = threadIdx.x;
  const size_t row = blockIdx.x;
  const float* xr = x + row * 1024;

  float v[4];
  float s = 0.f;
#pragma unroll
  for (int i = 0; i < 4; ++i) { v[i] = xr[t + i * 256]; s += v[i]; }
  red[t] = s;
  __syncthreads();
  for (int o = 128; o > 0; o >>= 1) {
    if (t < o) red[t] += red[t + o];
    __syncthreads();
  }
  const float mu = red[0] * (1.f / 1024.f);
  __syncthreads();

  float sq = 0.f;
#pragma unroll
  for (int i = 0; i < 4; ++i) { float d = v[i] - mu; sq += d * d; }
  red[t] = sq;
  __syncthreads();
  for (int o = 128; o > 0; o >>= 1) {
    if (t < o) red[t] += red[t + o];
    __syncthreads();
  }
  const float inv = rsqrtf(red[0] * (1.f / 1024.f) + 1e-5f);

#pragma unroll
  for (int i = 0; i < 4; ++i) {
    const int c = t + i * 256;
    out[row * 1024 + c] = (_Float16)((v[i] - mu) * inv * w[c] + b[c]);
  }
}

// ---------------------------------------------------------------------------
// WMMA GEMM: out[M,N] = A[M,K](f16) @ W[K,N] + bias (+gelu / +residual).
// A is f16 row-major; W is pre-packed by pack_weight_kernel.
// Block = 256 threads = 8 waves (2x4); block tile 128x256; wave tile 64x64.
// Tiles double-buffered in LDS via global_load_async_to_lds_b128:
//   6 async ops/wave/stage; s_wait_asynccnt 6 overlaps next-stage DMA with
//   compute of the current stage.
// EPI: 0 = bias -> f16 ; 1 = bias+tanh-GELU -> f16 ; 2 = bias+resid -> f32.
// M%128==0, N%256==0, K%32==0 (true for all calls here).
// ---------------------------------------------------------------------------
template <int EPI>
__global__ __launch_bounds__(256)
void gemm_kernel(const _Float16* __restrict__ A, const unsigned* __restrict__ Bpk,
                 const float* __restrict__ bias, const float* __restrict__ resid,
                 _Float16* __restrict__ outH, float* __restrict__ outF,
                 int M, int N, int K) {
  __shared__ _Float16 sA[2][128 * 32];  // [buf][row][k]            2 x 8 KB
  __shared__ unsigned sB[2][16 * 256];  // [buf][col][kpair] packed 2 x 16 KB

  const int t = threadIdx.x;
  const int lane = t & 31;
  const int wave = t >> 5;
  const int wm = wave >> 2;  // 0..1
  const int wn = wave & 3;   // 0..3
  const int mBase = blockIdx.y * 128;
  const int nBase = blockIdx.x * 256;

  // Issue one stage of async tile DMA (6 x b128 per thread/wave).
  auto stage = [&](int k0, int buf) {
#pragma unroll
    for (int i = 0; i < 2; ++i) {              // A: 128x32 halves = 8 KB
      const int dw = i * 1024 + t * 4;         // dword index in tile
      const int row = dw >> 4;
      const int pr = dw & 15;
      const void* g = (const void*)(A + (size_t)(mBase + row) * K + k0 + pr * 2);
      async_b128((unsigned)(uintptr_t)&sA[buf][dw * 2], g);
    }
    const unsigned* gB = Bpk + ((size_t)(k0 >> 5) * N + nBase) * 16;  // 16 KB contig
#pragma unroll
    for (int i = 0; i < 4; ++i) {
      const int dw = i * 1024 + t * 4;
      async_b128((unsigned)(uintptr_t)&sB[buf][dw], (const void*)(gB + dw));
    }
  };

  const v8f z = {0.f, 0.f, 0.f, 0.f, 0.f, 0.f, 0.f, 0.f};
  v8f acc[4][4];
#pragma unroll
  for (int mt = 0; mt < 4; ++mt)
#pragma unroll
    for (int nt = 0; nt < 4; ++nt) acc[mt][nt] = z;

  stage(0, 0);
  int cur = 0;
  for (int k0 = 0; k0 < K; k0 += 32, cur ^= 1) {
    if (k0 + 32 < K) {
      stage(k0 + 32, cur ^ 1);
      asm volatile("s_wait_asynccnt 6" ::: "memory");  // prev stage done
    } else {
      asm volatile("s_wait_asynccnt 0" ::: "memory");
    }
    __syncthreads();

    const unsigned* sAd = (const unsigned*)sA[cur];
    const unsigned* sBd = sB[cur];

    Frag16 aF[4];
#pragma unroll
    for (int mt = 0; mt < 4; ++mt) {
      const int row = wm * 64 + mt * 16 + (lane & 15);
#pragma unroll
      for (int i = 0; i < 8; ++i)
        aF[mt].u[i] = sAd[row * 16 + (aklo(i, lane) >> 1)];
    }
#pragma unroll
    for (int nt = 0; nt < 4; ++nt) {
      Frag16 bF;
      const int col = wn * 64 + nt * 16 + (lane & 15);
#pragma unroll
      for (int j = 0; j < 8; ++j) {
        const int pk = ((lane < 16) ? 0 : 8) + j;
        bF.u[j] = sBd[col * 16 + pk];  // 8 consecutive dwords -> 2x ds_load_b128
      }
#pragma unroll
      for (int mt = 0; mt < 4; ++mt)
        acc[mt][nt] = wmma_f16(aF[mt].v, bF.v, acc[mt][nt]);
    }
    __syncthreads();  // all waves done reading before buf is re-filled
  }

  // epilogue: C/D layout -> lane<16: M=r, else M=8+r; N = lane&15
#pragma unroll
  for (int mt = 0; mt < 4; ++mt) {
#pragma unroll
    for (int nt = 0; nt < 4; ++nt) {
#pragma unroll
      for (int r = 0; r < 8; ++r) {
        const int ml = wm * 64 + mt * 16 + ((lane < 16) ? r : 8 + r);
        const int nl = wn * 64 + nt * 16 + (lane & 15);
        const size_t gm = (size_t)(mBase + ml);
        const size_t gn = (size_t)(nBase + nl);
        float v = acc[mt][nt][r] + bias[gn];
        if (EPI == 1) {
          const float u = v;
          v = 0.5f * u *
              (1.f + tanhf(0.7978845608028654f * (u + 0.044715f * u * u * u)));
        }
        if (EPI == 2) {
          v += resid[gm * N + gn];
          outF[gm * N + gn] = v;
        } else {
          outH[gm * N + gn] = (_Float16)v;
        }
      }
    }
  }
}

// ---------------------------------------------------------------------------
// Causal attention: one wave (32 threads) per 16-query tile of one (b,h).
// qkv: [B*T, 3C] f16 (Q | K | V, head h at channel h*64).
// y  : [B*T, C]  f16 (channel = h*64 + d).
// Dynamic LDS: 16 rows x 1024 fp32 scores (64 KB), softmax'ed to f16 in-place.
// ---------------------------------------------------------------------------
__global__ __launch_bounds__(32)
void attention_kernel(const _Float16* __restrict__ qkv, _Float16* __restrict__ y) {
  extern __shared__ float sS[];  // 16*1024 floats
  const int lane = threadIdx.x;
  const int qt = blockIdx.x;     // 0..63 (query tile)
  const int h = blockIdx.y;      // 0..15
  const int b = blockIdx.z;      // 0..7
  const int T = 1024, C3 = 3072, C = 1024;
  const size_t base = (size_t)b * T * C3;
  const int m = lane & 15;

  // Q fragments: 16x64 split into two 16x32 A-fragments.
  Frag16 qF[2];
#pragma unroll
  for (int c = 0; c < 2; ++c)
#pragma unroll
    for (int i = 0; i < 8; ++i) {
      const int klo = aklo(i, lane);
      qF[c].u[i] = *(const unsigned*)(qkv + base + (size_t)(qt * 16 + m) * C3 +
                                      h * 64 + c * 32 + klo);
    }

  // ---- scores: S[16 x L] = (Q @ K^T) * 1/sqrt(64), causal-masked ----
  const int nkb = qt + 1;  // 16-key blocks (causal bound)
  for (int kb = 0; kb < nkb; ++kb) {
    Frag16 kF[2];
    const int keyTok = kb * 16 + (lane & 15);  // B-frag N = key
#pragma unroll
    for (int c = 0; c < 2; ++c)
#pragma unroll
      for (int j = 0; j < 8; ++j) {
        const int krow = ((lane < 16) ? 0 : 16) + 2 * j;  // B-frag K = d
        kF[c].u[j] = *(const unsigned*)(qkv + base + (size_t)keyTok * C3 + C +
                                        h * 64 + c * 32 + krow);
      }
    v8f s = {0.f, 0.f, 0.f, 0.f, 0.f, 0.f, 0.f, 0.f};
    s = wmma_f16(qF[0].v, kF[0].v, s);
    s = wmma_f16(qF[1].v, kF[1].v, s);
#pragma unroll
    for (int r = 0; r < 8; ++r) {
      const int mr = (lane < 16) ? r : 8 + r;
      const int key = kb * 16 + (lane & 15);
      const float val = (key <= qt * 16 + mr) ? s[r] * 0.125f : -1e30f;
      sS[mr * 1024 + key] = val;
    }
  }
  __syncthreads();

  // ---- softmax per row, in place; f16 P written over its own row ----
  const int L = nkb * 16;
  const int Lp = (L + 31) & ~31;
  if (lane < 16) {
    float* row = sS + lane * 1024;
    float mx = -1e30f;
    for (int c = 0; c < L; ++c) mx = fmaxf(mx, row[c]);
    float sum = 0.f;
    for (int c = 0; c < L; ++c) { const float e = __expf(row[c] - mx); row[c] = e; sum += e; }
    const float inv = 1.f / sum;
    _Float16* prow = (_Float16*)sS + (size_t)lane * 2048;
    for (int c = 0; c < L; ++c) prow[c] = (_Float16)(row[c] * inv);  // ascending: in-place ok
    for (int c = L; c < Lp; ++c) prow[c] = (_Float16)0.f;            // zero-pad to 32
  }
  __syncthreads();

  // ---- O[16 x 64] = P[16 x Lp] @ V[Lp x 64] ----
  const v8f z = {0.f, 0.f, 0.f, 0.f, 0.f, 0.f, 0.f, 0.f};
  v8f o[4];
#pragma unroll
  for (int nt = 0; nt < 4; ++nt) o[nt] = z;
  const unsigned* sSd = (const unsigned*)sS;

  for (int kc = 0; kc < Lp; kc += 32) {
    Frag16 pF;  // A-fragment from LDS P (row stride 2048 halves = 1024 dwords)
#pragma unroll
    for (int i = 0; i < 8; ++i) {
      const int klo = aklo(i, lane);
      pF.u[i] = sSd[m * 1024 + ((kc + klo) >> 1)];
    }
#pragma unroll
    for (int nt = 0; nt < 4; ++nt) {
      Frag16 vF;
      const int d = nt * 16 + (lane & 15);  // B-frag N = head dim
#pragma unroll
      for (int j = 0; j < 8; ++j) {
        const int tok = kc + ((lane < 16) ? 0 : 16) + 2 * j;
        const unsigned lo = *(const unsigned short*)(qkv + base + (size_t)tok * C3 + 2 * C + h * 64 + d);
        const unsigned hi = *(const unsigned short*)(qkv + base + (size_t)(tok + 1) * C3 + 2 * C + h * 64 + d);
        vF.u[j] = lo | (hi << 16);
      }
      o[nt] = wmma_f16(pF.v, vF.v, o[nt]);
    }
  }

#pragma unroll
  for (int nt = 0; nt < 4; ++nt)
#pragma unroll
    for (int r = 0; r < 8; ++r) {
      const int mr = (lane < 16) ? r : 8 + r;
      y[((size_t)b * T + qt * 16 + mr) * 1024 + h * 64 + nt * 16 + (lane & 15)] =
          (_Float16)o[nt][r];
    }
}

// ---------------------------------------------------------------------------
// Host-side orchestration
// ---------------------------------------------------------------------------
extern "C" void kernel_launch(void* const* d_in, const int* in_sizes, int n_in,
                              void* d_out, int out_size, void* d_ws, size_t ws_size,
                              hipStream_t stream) {
  (void)in_sizes; (void)n_in; (void)out_size; (void)ws_size;
  const float* x      = (const float*)d_in[0];
  const float* ln1_w  = (const float*)d_in[1];
  const float* ln1_b  = (const float*)d_in[2];
  const float* w_qkv  = (const float*)d_in[3];
  const float* b_qkv  = (const float*)d_in[4];
  const float* w_o    = (const float*)d_in[5];
  const float* b_o    = (const float*)d_in[6];
  const float* ln2_w  = (const float*)d_in[7];
  const float* ln2_b  = (const float*)d_in[8];
  const float* w_fc   = (const float*)d_in[9];
  const float* b_fc   = (const float*)d_in[10];
  const float* w_proj = (const float*)d_in[11];
  const float* b_proj = (const float*)d_in[12];

  // Workspace layout (~200 MB)
  const size_t MB = 1024ull * 1024ull;
  char* ws = (char*)d_ws;
  _Float16* xn   = (_Float16*)(ws + 0);         // [8192,1024] f16, reused LN1/LN2
  _Float16* qkvB = (_Float16*)(ws + 16 * MB);   // [8192,3072] f16
  _Float16* yB   = (_Float16*)(ws + 64 * MB);   // [8192,1024] f16
  float*    x1   = (float*)   (ws + 80 * MB);   // [8192,1024] f32 residual
  _Float16* hB   = (_Float16*)(ws + 112 * MB);  // [8192,4096] f16
  unsigned* wqP  = (unsigned*)(ws + 176 * MB);  // packed [1024,3072] f16
  unsigned* woP  = (unsigned*)(ws + 182 * MB);  // packed [1024,1024] f16
  unsigned* wfcP = (unsigned*)(ws + 184 * MB);  // packed [1024,4096] f16
  unsigned* wprP = (unsigned*)(ws + 192 * MB);  // packed [4096,1024] f16

  const int Mrows = 8192;  // B*T

  // Weight downconversion + WMMA-native packing (deterministic each call).
  pack_weight_kernel<<<2048, 256, 0, stream>>>(w_qkv, wqP, 1024, 3072);
  pack_weight_kernel<<<2048, 256, 0, stream>>>(w_o, woP, 1024, 1024);
  pack_weight_kernel<<<2048, 256, 0, stream>>>(w_fc, wfcP, 1024, 4096);
  pack_weight_kernel<<<2048, 256, 0, stream>>>(w_proj, wprP, 4096, 1024);

  // x -> LN1 -> f16
  layernorm_f16_kernel<<<Mrows, 256, 0, stream>>>(x, ln1_w, ln1_b, xn);
  // qkv = xn @ w_qkv + b_qkv
  gemm_kernel<0><<<dim3(3072 / 256, Mrows / 128), 256, 0, stream>>>(
      xn, wqP, b_qkv, nullptr, qkvB, nullptr, Mrows, 3072, 1024);
  // causal attention -> y (f16)
  attention_kernel<<<dim3(64, 16, 8), 32, 16 * 1024 * sizeof(float), stream>>>(qkvB, yB);
  // x1 = x + y @ w_o + b_o  (fp32 residual stream)
  gemm_kernel<2><<<dim3(1024 / 256, Mrows / 128), 256, 0, stream>>>(
      yB, woP, b_o, x, nullptr, x1, Mrows, 1024, 1024);
  // LN2 -> f16
  layernorm_f16_kernel<<<Mrows, 256, 0, stream>>>(x1, ln2_w, ln2_b, xn);
  // h = gelu(xn @ w_fc + b_fc)
  gemm_kernel<1><<<dim3(4096 / 256, Mrows / 128), 256, 0, stream>>>(
      xn, wfcP, b_fc, nullptr, hB, nullptr, Mrows, 4096, 1024);
  // out = x1 + h @ w_proj + b_proj
  gemm_kernel<2><<<dim3(1024 / 256, Mrows / 128), 256, 0, stream>>>(
      hB, wprP, b_proj, x1, nullptr, (float*)d_out, Mrows, 1024, 4096);
}
